// TensorProductAttention_60172491817301
// MI455X (gfx1250) — matrix operand order repaired and checked by reading, not compiled, not run on hardware
//
#include <hip/hip_runtime.h>
#include <hip/hip_bf16.h>

// ---------------------------------------------------------------------------
// Tensor Product Attention for MI455X (gfx1250, wave32, WMMA)
//   D_MODEL=768, H=12, DK=64, RQ=6, RK=2, RV=2, B=8, S=4096  -> N = 32768 tokens
//
// Pipeline:
//   prep_kernel      : f32 weights -> f16, transposed [n][k]; fused bias vec
//   proj_attn_kernel : X @ Wcat (WMMA f16, double-buffered K loop) -> P tile
//                      in LDS -> low-rank head-attention epilogue -> Y (f16,
//                      scrambled layout matching reference transpose+reshape)
//   out_gemm_kernel  : Y @ Wout (WMMA f16, double-buffered) + b_out -> f32
// ---------------------------------------------------------------------------

typedef __attribute__((ext_vector_type(16))) _Float16 v16h;
typedef __attribute__((ext_vector_type(8)))  float    v8f;

#define DM      768
#define NTOK    32768
#define SEQ     4096
#define PLD     776          // LDS row stride (776 % 64 != 0 -> no bank conflict)

// column offsets inside the fused projection output (total 760, padded to 768)
#define OFF_AQ  0            // 72  = RQ*H   (r*12+h)
#define OFF_BQ  72           // 384 = RQ*DK  (r*64+d)
#define OFF_AK  456          // 24  = RK*H
#define OFF_BK  480          // 128 = RK*DK
#define OFF_AV  608          // 24
#define OFF_BV  632          // 128

union FragU { uint4 u[2]; v16h h; };

// A-fragment (16x32 f16, row-major source, row = lane%16):
//   VGPR 0..7 hold contiguous K = 16*half + [0,16)
__device__ inline v16h load_a_f16(const _Float16* __restrict__ row, int kk, int half) {
  FragU f;
  f.u[0] = *(const uint4*)(row + kk + 16 * half);
  f.u[1] = *(const uint4*)(row + kk + 16 * half + 8);
  return f.h;
}

__device__ inline v16h load_a_f32cvt(const float* __restrict__ row, int kk, int half) {
  int k0 = kk + 16 * half;
  int k1 = kk + 16 * half + 8;
  float4 f0 = *(const float4*)(row + k0);
  float4 f1 = *(const float4*)(row + k0 + 4);
  float4 f2 = *(const float4*)(row + k1);
  float4 f3 = *(const float4*)(row + k1 + 4);
  v16h a;
  a[0]  = (_Float16)f0.x; a[1]  = (_Float16)f0.y; a[2]  = (_Float16)f0.z; a[3]  = (_Float16)f0.w;
  a[4]  = (_Float16)f1.x; a[5]  = (_Float16)f1.y; a[6]  = (_Float16)f1.z; a[7]  = (_Float16)f1.w;
  a[8]  = (_Float16)f2.x; a[9]  = (_Float16)f2.y; a[10] = (_Float16)f2.z; a[11] = (_Float16)f2.w;
  a[12] = (_Float16)f3.x; a[13] = (_Float16)f3.y; a[14] = (_Float16)f3.z; a[15] = (_Float16)f3.w;
  return a;
}

// B-fragment (32x16 f16): lane holds column n = lane%16, contiguous K range
// [16*half, 16*half+16). `row` already points at wt + n*DM + 16*half.
__device__ inline v16h load_b_f16(const _Float16* __restrict__ row, int kk) {
  FragU f;
  f.u[0] = ((const uint4*)(row + kk))[0];
  f.u[1] = ((const uint4*)(row + kk))[1];
  return f.h;
}

// ---------------------------------------------------------------------------
// Weight prep: fused projection weight (transposed, f16), W_out (transposed,
// f16), fused projection bias (f32).
// ---------------------------------------------------------------------------
__global__ void prep_kernel(const float* __restrict__ Waq, const float* __restrict__ baq,
                            const float* __restrict__ Wbq, const float* __restrict__ bbq,
                            const float* __restrict__ Wak, const float* __restrict__ bak,
                            const float* __restrict__ Wbk, const float* __restrict__ bbk,
                            const float* __restrict__ Wav, const float* __restrict__ bav,
                            const float* __restrict__ Wbv, const float* __restrict__ bbv,
                            const float* __restrict__ Wout,
                            _Float16* __restrict__ wcat, _Float16* __restrict__ woutt,
                            float* __restrict__ bcat) {
  int idx = blockIdx.x * 256 + threadIdx.x;
  if (idx >= DM * DM) return;
  int n = idx / DM;
  int k = idx - n * DM;
  float v;
  if      (n < OFF_BQ)      v = Waq[k * 72  + n];
  else if (n < OFF_AK)      v = Wbq[k * 384 + (n - OFF_BQ)];
  else if (n < OFF_BK)      v = Wak[k * 24  + (n - OFF_AK)];
  else if (n < OFF_AV)      v = Wbk[k * 128 + (n - OFF_BK)];
  else if (n < OFF_BV)      v = Wav[k * 24  + (n - OFF_AV)];
  else if (n < 760)         v = Wbv[k * 128 + (n - OFF_BV)];
  else                      v = 0.0f;
  wcat[idx]  = (_Float16)v;                  // idx == n*768 + k
  woutt[idx] = (_Float16)Wout[k * DM + n];   // transpose
  if (idx < DM) {
    float bv;
    if      (idx < OFF_BQ) bv = baq[idx];
    else if (idx < OFF_AK) bv = bbq[idx - OFF_BQ];
    else if (idx < OFF_BK) bv = bak[idx - OFF_AK];
    else if (idx < OFF_AV) bv = bbk[idx - OFF_BK];
    else if (idx < OFF_BV) bv = bav[idx - OFF_AV];
    else if (idx < 760)    bv = bbv[idx - OFF_BV];
    else                   bv = 0.0f;
    bcat[idx] = bv;
  }
}

// ---------------------------------------------------------------------------
// Fused projection GEMM + low-rank head attention.
// Block: 16 tokens x 768 cols, 256 threads = 8 waves, 6 N-tiles / wave.
// K loop double-buffered so loads of stage i+1 overlap the 6 WMMAs of stage i.
// ---------------------------------------------------------------------------
__global__ __launch_bounds__(256) void proj_attn_kernel(
    const float* __restrict__ x, const float* __restrict__ rope,
    const _Float16* __restrict__ wcat, const float* __restrict__ bcat,
    _Float16* __restrict__ Y) {
  __shared__ float P[16 * PLD];   // per-token projection rows (with bias)
  __shared__ float G[16 * 12];    // per-token Gram factors G[r1][r2]

  const int tid  = threadIdx.x;
  const int lane = tid & 31;
  const int wave = tid >> 5;
  const int half = lane >> 4;
  const int l16  = lane & 15;
  const int m_base = blockIdx.x * 16;

  // ---- GEMM phase: P = X @ Wcat ------------------------------------------
  v8f acc[6];
#pragma unroll
  for (int i = 0; i < 6; ++i)
#pragma unroll
    for (int j = 0; j < 8; ++j) acc[i][j] = 0.0f;

  const float* xrow = x + (size_t)(m_base + l16) * DM;
  const _Float16* brow[6];
#pragma unroll
  for (int nt = 0; nt < 6; ++nt)
    brow[nt] = wcat + (size_t)(wave * 96 + nt * 16 + l16) * DM + 16 * half;

  v16h a0 = load_a_f32cvt(xrow, 0, half);
  v16h b0[6], b1[6];
#pragma unroll
  for (int nt = 0; nt < 6; ++nt) b0[nt] = load_b_f16(brow[nt], 0);

  for (int kk = 0; kk < DM; kk += 64) {
    // stage 1 prefetch (kk+32 always < DM since DM % 64 == 0)
    v16h a1 = load_a_f32cvt(xrow, kk + 32, half);
#pragma unroll
    for (int nt = 0; nt < 6; ++nt) b1[nt] = load_b_f16(brow[nt], kk + 32);
#pragma unroll
    for (int nt = 0; nt < 6; ++nt)
      acc[nt] = __builtin_amdgcn_wmma_f32_16x16x32_f16(
          false, a0, false, b0[nt], (short)0, acc[nt], false, false);

    // stage 0 prefetch for next iteration
    if (kk + 64 < DM) {
      a0 = load_a_f32cvt(xrow, kk + 64, half);
#pragma unroll
      for (int nt = 0; nt < 6; ++nt) b0[nt] = load_b_f16(brow[nt], kk + 64);
    }
#pragma unroll
    for (int nt = 0; nt < 6; ++nt)
      acc[nt] = __builtin_amdgcn_wmma_f32_16x16x32_f16(
          false, a1, false, b1[nt], (short)0, acc[nt], false, false);
  }

  // C/D layout: VGPR v -> (m = v + 8*half, n = lane%16). Add bias, stash LDS.
#pragma unroll
  for (int nt = 0; nt < 6; ++nt) {
    int n = wave * 96 + nt * 16 + l16;
    float bias = bcat[n];
#pragma unroll
    for (int v = 0; v < 8; ++v) {
      int m = v + 8 * half;
      P[m * PLD + n] = acc[nt][v] + bias;
    }
  }
  __syncthreads();

  // ---- Phase A: Gram factors G[r1][r2] = sum_d Bq*Bk*rope^2 --------------
  if (tid < 192) {
    int t = tid / 12, idx = tid % 12;
    int r1 = idx >> 1, r2 = idx & 1;
    int tok = m_base + t;
    int s = tok & (SEQ - 1);
    const float* rp = rope + (size_t)s * 64;
    const float* pr = P + t * PLD;
    float g = 0.0f;
#pragma unroll 8
    for (int d = 0; d < 64; ++d) {
      float ro = rp[d];
      g += pr[OFF_BQ + r1 * 64 + d] * pr[OFF_BK + r2 * 64 + d] * ro * ro;
    }
    G[t * 12 + r1 * 2 + r2] = g;
  }
  __syncthreads();

  // ---- Phase B: per-(token, head) scores / softmax / output --------------
  if (tid < 192) {
    int t = tid / 12, h = tid % 12;
    int tok = m_base + t;
    const float* pr = P + t * PLD;

    float c0 = 0.0f, c1 = 0.0f;
#pragma unroll
    for (int r1 = 0; r1 < 6; ++r1) {
      float aq = pr[OFF_AQ + r1 * 12 + h];
      c0 += aq * G[t * 12 + r1 * 2 + 0];
      c1 += aq * G[t * 12 + r1 * 2 + 1];
    }
    float sc[12], mx = -3.0e38f;
#pragma unroll
    for (int g = 0; g < 12; ++g) {
      float s = 0.125f * (pr[OFF_AK + g] * c0 + pr[OFF_AK + 12 + g] * c1);
      sc[g] = s;
      mx = fmaxf(mx, s);
    }
    float sum = 0.0f;
#pragma unroll
    for (int g = 0; g < 12; ++g) { sc[g] = __expf(sc[g] - mx); sum += sc[g]; }
    float inv = 1.0f / sum;
    float w0 = 0.0f, w1 = 0.0f;
#pragma unroll
    for (int g = 0; g < 12; ++g) {
      float a = sc[g] * inv;
      w0 += a * pr[OFF_AV + g];
      w1 += a * pr[OFF_AV + 12 + g];
    }
    // Reference does transpose(0,2,1,3).reshape(b,s,768): flat offset within
    // batch b is h*S*64 + s*64 + d. Write Y in that (scrambled) layout so the
    // final GEMM sees a plain row-major [32768 x 768] matrix.
    int b = tok >> 12;          // tok / SEQ
    int s = tok & (SEQ - 1);
    _Float16* yrow = Y + (size_t)b * (SEQ * DM) + (size_t)h * (SEQ * 64) + (size_t)s * 64;
#pragma unroll 8
    for (int d = 0; d < 64; ++d) {
      float o = w0 * pr[OFF_BV + d] + w1 * pr[OFF_BV + 64 + d];
      yrow[d] = (_Float16)o;
    }
  }
}

// ---------------------------------------------------------------------------
// Output GEMM: out = Y @ W_out + b_out  (f16 WMMA, f32 out, double-buffered)
// ---------------------------------------------------------------------------
__global__ __launch_bounds__(256) void out_gemm_kernel(
    const _Float16* __restrict__ Y, const _Float16* __restrict__ woutt,
    const float* __restrict__ bout, float* __restrict__ out) {
  const int tid  = threadIdx.x;
  const int lane = tid & 31;
  const int wave = tid >> 5;
  const int half = lane >> 4;
  const int l16  = lane & 15;
  const int m_base = blockIdx.x * 16;

  v8f acc[6];
#pragma unroll
  for (int i = 0; i < 6; ++i)
#pragma unroll
    for (int j = 0; j < 8; ++j) acc[i][j] = 0.0f;

  const _Float16* yrow = Y + (size_t)(m_base + l16) * DM;
  const _Float16* brow[6];
#pragma unroll
  for (int nt = 0; nt < 6; ++nt)
    brow[nt] = woutt + (size_t)(wave * 96 + nt * 16 + l16) * DM + 16 * half;

  v16h a0 = load_a_f16(yrow, 0, half);
  v16h b0[6], b1[6];
#pragma unroll
  for (int nt = 0; nt < 6; ++nt) b0[nt] = load_b_f16(brow[nt], 0);

  for (int kk = 0; kk < DM; kk += 64) {
    v16h a1 = load_a_f16(yrow, kk + 32, half);
#pragma unroll
    for (int nt = 0; nt < 6; ++nt) b1[nt] = load_b_f16(brow[nt], kk + 32);
#pragma unroll
    for (int nt = 0; nt < 6; ++nt)
      acc[nt] = __builtin_amdgcn_wmma_f32_16x16x32_f16(
          false, a0, false, b0[nt], (short)0, acc[nt], false, false);

    if (kk + 64 < DM) {
      a0 = load_a_f16(yrow, kk + 64, half);
#pragma unroll
      for (int nt = 0; nt < 6; ++nt) b0[nt] = load_b_f16(brow[nt], kk + 64);
    }
#pragma unroll
    for (int nt = 0; nt < 6; ++nt)
      acc[nt] = __builtin_amdgcn_wmma_f32_16x16x32_f16(
          false, a1, false, b1[nt], (short)0, acc[nt], false, false);
  }

#pragma unroll
  for (int nt = 0; nt < 6; ++nt) {
    int n = wave * 96 + nt * 16 + l16;
    float bias = bout[n];
#pragma unroll
    for (int v = 0; v < 8; ++v) {
      int m = m_base + v + 8 * half;
      out[(size_t)m * DM + n] = acc[nt][v] + bias;
    }
  }
}

// ---------------------------------------------------------------------------
extern "C" void kernel_launch(void* const* d_in, const int* in_sizes, int n_in,
                              void* d_out, int out_size, void* d_ws, size_t ws_size,
                              hipStream_t stream) {
  const float* x    = (const float*)d_in[0];
  const float* rope = (const float*)d_in[1];
  const float* Waq  = (const float*)d_in[2];
  const float* baq  = (const float*)d_in[3];
  const float* Wbq  = (const float*)d_in[4];
  const float* bbq  = (const float*)d_in[5];
  const float* Wak  = (const float*)d_in[6];
  const float* bak  = (const float*)d_in[7];
  const float* Wbk  = (const float*)d_in[8];
  const float* bbk  = (const float*)d_in[9];
  const float* Wav  = (const float*)d_in[10];
  const float* bav  = (const float*)d_in[11];
  const float* Wbv  = (const float*)d_in[12];
  const float* bbv  = (const float*)d_in[13];
  const float* Wout = (const float*)d_in[14];
  const float* bout = (const float*)d_in[15];

  // Workspace layout (all 256B-aligned):
  //   wcat  f16 [768][768]  : 1,179,648 B  @ 0
  //   woutt f16 [768][768]  : 1,179,648 B  @ 1,179,648
  //   bcat  f32 [768]       :     3,072 B  @ 2,359,296
  //   Y     f16 [32768][768]: 50,331,648 B @ 2,362,368   (total ~52.7 MB)
  char* ws = (char*)d_ws;
  _Float16* wcat  = (_Float16*)(ws);
  _Float16* woutt = (_Float16*)(ws + 1179648);
  float*    bcat  = (float*)(ws + 2359296);
  _Float16* Y     = (_Float16*)(ws + 2362368);

  prep_kernel<<<(DM * DM + 255) / 256, 256, 0, stream>>>(
      Waq, baq, Wbq, bbq, Wak, bak, Wbk, bbk, Wav, bav, Wbv, bbv,
      Wout, wcat, woutt, bcat);

  proj_attn_kernel<<<NTOK / 16, 256, 0, stream>>>(x, rope, wcat, bcat, Y);

  out_gemm_kernel<<<NTOK / 16, 256, 0, stream>>>(Y, woutt, bout, (float*)d_out);
}
